// HVSA_32650341384488
// MI455X (gfx1250) — matrix operand order repaired and checked by reading, not compiled, not run on hardware
//
#include <hip/hip_runtime.h>
#include <math.h>

typedef __bf16 bf16;
typedef __attribute__((ext_vector_type(4)))  __bf16 v4bf;
typedef __attribute__((ext_vector_type(8)))  __bf16 v8bf;
typedef __attribute__((ext_vector_type(16))) __bf16 v16bf;
typedef __attribute__((ext_vector_type(8)))  float  v8f;

#define HVSA_EPS 1e-8f

// ---------------------------------------------------------------------------
// Kernel 1: per-row preprocessing (one wave32 per row).
//   - split fp32 -> bf16 hi + bf16 lo (residual) for error-corrected WMMA
//   - compute row L2 norm via wave32 shuffle reduction
// ---------------------------------------------------------------------------
__global__ __launch_bounds__(256) void hvsa_prep(
    const float* __restrict__ img, const float* __restrict__ txt,
    bf16* __restrict__ imgHi, bf16* __restrict__ imgLo,
    bf16* __restrict__ txtHi, bf16* __restrict__ txtLo,
    float* __restrict__ normImg, float* __restrict__ normTxt,
    int Bimg, int Btxt, int D)
{
    const int wid  = (int)((blockIdx.x * blockDim.x + threadIdx.x) >> 5);
    const int lane = (int)(threadIdx.x & 31);
    const int rows = Bimg + Btxt;
    if (wid >= rows) return;

    const float* src;
    bf16 *dhi, *dlo;
    float* nrm;
    if (wid < Bimg) {
        src = img  + (size_t)wid * D;
        dhi = imgHi + (size_t)wid * D;
        dlo = imgLo + (size_t)wid * D;
        nrm = normImg + wid;
    } else {
        const int r = wid - Bimg;
        src = txt  + (size_t)r * D;
        dhi = txtHi + (size_t)r * D;
        dlo = txtLo + (size_t)r * D;
        nrm = normTxt + r;
    }

    float ssq = 0.0f;
    for (int k = lane * 4; k + 3 < D; k += 128) {
        const float4 v = *reinterpret_cast<const float4*>(src + k);
        float f[4] = {v.x, v.y, v.z, v.w};
        v4bf h, l;
#pragma unroll
        for (int i = 0; i < 4; ++i) {
            const bf16 hh = (bf16)f[i];
            h[i] = hh;
            l[i] = (bf16)(f[i] - (float)hh);
            ssq  = fmaf(f[i], f[i], ssq);
        }
        *reinterpret_cast<v4bf*>(dhi + k) = h;
        *reinterpret_cast<v4bf*>(dlo + k) = l;
    }

#pragma unroll
    for (int off = 16; off > 0; off >>= 1)
        ssq += __shfl_xor(ssq, off, 32);
    if (lane == 0) *nrm = sqrtf(ssq);
}

// ---------------------------------------------------------------------------
// Fragment loaders for WMMA 16-bit operands (wave32); A and B share this code's contiguous K order.
// A (16x32, MxK): lane l holds row M=l&15; contiguous K {(l>>4)*16 .. +15}.
// B (32x16, KxN): lane l holds column N=l&15; contiguous K {(l>>4)*16 .. +15}.
// ---------------------------------------------------------------------------
__device__ __forceinline__ v16bf load_a_frag(const bf16* __restrict__ p) {
    const v8bf a = *reinterpret_cast<const v8bf*>(p);
    const v8bf b = *reinterpret_cast<const v8bf*>(p + 8);
    return __builtin_shufflevector(a, b, 0, 1, 2, 3, 4, 5, 6, 7,
                                         8, 9, 10, 11, 12, 13, 14, 15);
}
__device__ __forceinline__ v16bf load_b_frag(const bf16* __restrict__ p) {
    return *reinterpret_cast<const v16bf*>(p);
}

// ---------------------------------------------------------------------------
// Kernel 2: bf16x3 error-corrected WMMA GEMM + cosine normalization epilogue.
// Block = 256 threads = 8 waves; block tile 128x128; wave tile 32x64
// (2x4 grid of 16x16 f32 accumulators). NT GEMM: both operands row-major,
// contraction over D, so A and B fragments both come from row-major rows.
// ---------------------------------------------------------------------------
__global__ __launch_bounds__(256) void hvsa_gemm(
    const bf16* __restrict__ imgHi, const bf16* __restrict__ imgLo,
    const bf16* __restrict__ txtHi, const bf16* __restrict__ txtLo,
    const float* __restrict__ normImg, const float* __restrict__ normTxt,
    float* __restrict__ out, int Bimg, int Btxt, int D)
{
    const int lane = (int)(threadIdx.x & 31);
    const int wave = (int)(threadIdx.x >> 5);
    const int wm   = wave >> 1;                  // 0..3
    const int wn   = wave & 1;                   // 0..1
    const int m0   = (int)blockIdx.y * 128 + wm * 32;
    const int n0   = (int)blockIdx.x * 128 + wn * 64;

    const int mrow  = lane & 15;
    const int hi16  = lane >> 4;
    const int koffA = hi16 * 16;
    const int koffB = hi16 * 16;

    v8f acc[2][4];
#pragma unroll
    for (int mi = 0; mi < 2; ++mi)
#pragma unroll
        for (int ni = 0; ni < 4; ++ni)
            acc[mi][ni] = (v8f){0.f, 0.f, 0.f, 0.f, 0.f, 0.f, 0.f, 0.f};

    for (int k0 = 0; k0 < D; k0 += 32) {
        v16bf aHi[2], aLo[2], bHi[4], bLo[4];
#pragma unroll
        for (int mi = 0; mi < 2; ++mi) {
            int r = m0 + mi * 16 + mrow;
            r = (r < Bimg) ? r : (Bimg - 1);         // clamp keeps EXEC full
            const size_t off = (size_t)r * D + k0 + koffA;
            aHi[mi] = load_a_frag(imgHi + off);
            aLo[mi] = load_a_frag(imgLo + off);
        }
#pragma unroll
        for (int ni = 0; ni < 4; ++ni) {
            int c = n0 + ni * 16 + mrow;
            c = (c < Btxt) ? c : (Btxt - 1);
            const size_t off = (size_t)c * D + k0 + koffB;
            bHi[ni] = load_b_frag(txtHi + off);
            bLo[ni] = load_b_frag(txtLo + off);
        }
        // bf16x3: dot(hi+lo, hi+lo) ~= hi*lo + lo*hi + hi*hi  (drop lo*lo)
#pragma unroll
        for (int mi = 0; mi < 2; ++mi)
#pragma unroll
            for (int ni = 0; ni < 4; ++ni) {
                acc[mi][ni] = __builtin_amdgcn_wmma_f32_16x16x32_bf16(
                    false, aHi[mi], false, bLo[ni], (short)0, acc[mi][ni], false, false);
                acc[mi][ni] = __builtin_amdgcn_wmma_f32_16x16x32_bf16(
                    false, aLo[mi], false, bHi[ni], (short)0, acc[mi][ni], false, false);
                acc[mi][ni] = __builtin_amdgcn_wmma_f32_16x16x32_bf16(
                    false, aHi[mi], false, bHi[ni], (short)0, acc[mi][ni], false, false);
            }
    }

    // Epilogue: C/D layout — VGPR r: lanes 0-15 -> M=r, lanes 16-31 -> M=r+8;
    // N = lane&15. Stores are 64B-contiguous across each half-wave.
    const int colN    = lane & 15;
    const int rowHalf = hi16 * 8;
#pragma unroll
    for (int ni = 0; ni < 4; ++ni) {
        const int N  = n0 + ni * 16 + colN;
        const float nt = (N < Btxt) ? normTxt[N] : 1.0f;
#pragma unroll
        for (int mi = 0; mi < 2; ++mi) {
#pragma unroll
            for (int r8 = 0; r8 < 8; ++r8) {
                const int M = m0 + mi * 16 + rowHalf + r8;
                if (M < Bimg && N < Btxt) {
                    const float nm    = normImg[M];
                    const float denom = fmaxf(nm * nt, HVSA_EPS);
                    out[(size_t)M * Btxt + N] = acc[mi][ni][r8] / denom;
                }
            }
        }
    }
}

// ---------------------------------------------------------------------------
extern "C" void kernel_launch(void* const* d_in, const int* in_sizes, int n_in,
                              void* d_out, int out_size, void* d_ws, size_t ws_size,
                              hipStream_t stream)
{
    const float* img = (const float*)d_in[0];
    const float* txt = (const float*)d_in[1];
    // d_in[2] (text_lens) is unused by the reference math.

    const int Btxt = in_sizes[2];
    const int Bimg = out_size / Btxt;
    const int D    = in_sizes[0] / Bimg;

    const size_t szImg = (size_t)Bimg * D;
    const size_t szTxt = (size_t)Btxt * D;

    bf16* imgHi = (bf16*)d_ws;
    bf16* imgLo = imgHi + szImg;
    bf16* txtHi = imgLo + szImg;
    bf16* txtLo = txtHi + szTxt;
    float* normImg = (float*)(txtLo + szTxt);
    float* normTxt = normImg + Bimg;

    const int rows = Bimg + Btxt;                 // one wave32 per row
    const int prepBlocks = (rows + 7) / 8;        // 8 waves per 256-thread block
    hipLaunchKernelGGL(hvsa_prep, dim3(prepBlocks), dim3(256), 0, stream,
                       img, txt, imgHi, imgLo, txtHi, txtLo,
                       normImg, normTxt, Bimg, Btxt, D);

    dim3 grid((Btxt + 127) / 128, (Bimg + 127) / 128);
    hipLaunchKernelGGL(hvsa_gemm, grid, dim3(256), 0, stream,
                       imgHi, imgLo, txtHi, txtLo, normImg, normTxt,
                       (float*)d_out, Bimg, Btxt, D);
}
